// TopKSAE_27152783245802
// MI455X (gfx1250) — matrix operand order repaired and checked by reading, not compiled, not run on hardware
//
#include <hip/hip_runtime.h>
#include <hip/hip_bf16.h>

#define D_MODEL 768
#define D_SAE   24576
#define BATCH   8192
#define TOPK    32
#define CAND    512

typedef __attribute__((ext_vector_type(16))) __bf16 bf16x16;
typedef __attribute__((ext_vector_type(8)))  float  f32x8;

union FragAB { bf16x16 v; uint4 u[2]; };

__device__ __forceinline__ unsigned short f2bf(float f) {
  unsigned int u = __float_as_uint(f);
  u += 0x7FFFu + ((u >> 16) & 1u);          // round-to-nearest-even
  return (unsigned short)(u >> 16);
}

// monotonic float -> uint key (ascending order preserved)
__device__ __forceinline__ unsigned int fkey(float f) {
  unsigned int u = __float_as_uint(f);
  return (u & 0x80000000u) ? ~u : (u | 0x80000000u);
}

// LDS byte offset of a __shared__ object (addrspace(3) pointers are 32-bit offsets)
__device__ __forceinline__ unsigned lds_off(const void* p) {
  return (unsigned)(unsigned long long)(__attribute__((address_space(3))) const void*)p;
}

// CDNA5 async global->LDS copy, 16B per lane, ASYNCcnt-tracked (GVS mode)
#define ASYNC_LOAD_B128(ldsoff, voff, sbase)                       \
  asm volatile("global_load_async_to_lds_b128 %0, %1, %2"          \
               :: "v"(ldsoff), "v"(voff), "s"(sbase) : "memory")
#define WAIT_ASYNC(n) asm volatile("s_wait_asynccnt %0" :: "i"(n) : "memory")

// ---------------------------------------------------------------------------
// Prep: f32 -> bf16 streaming convert (8 elems/thread)
// ---------------------------------------------------------------------------
__global__ __launch_bounds__(256)
void cvt_bf16(const float* __restrict__ src, unsigned short* __restrict__ dst, int n) {
  int i = (blockIdx.x * 256 + threadIdx.x) * 8;
  if (i + 7 < n) {
    float4 a = *(const float4*)(src + i);
    float4 b = *(const float4*)(src + i + 4);
    ushort4 o0 = { f2bf(a.x), f2bf(a.y), f2bf(a.z), f2bf(a.w) };
    ushort4 o1 = { f2bf(b.x), f2bf(b.y), f2bf(b.z), f2bf(b.w) };
    *(ushort4*)(dst + i)     = o0;
    *(ushort4*)(dst + i + 4) = o1;
  }
}

// ---------------------------------------------------------------------------
// Prep: W_dec [768][24576] -> WdT [24576][768] (coalesced decode gathers)
// ---------------------------------------------------------------------------
__global__ __launch_bounds__(256)
void wdec_transpose(const float* __restrict__ Wdec, float* __restrict__ WdT) {
  __shared__ float tile[32][33];
  const int tx = threadIdx.x & 31;
  const int ty = threadIdx.x >> 5;
  const size_t s0 = (size_t)blockIdx.x * 32;
  const size_t m0 = (size_t)blockIdx.y * 32;
#pragma unroll
  for (int j = 0; j < 32; j += 8)
    tile[ty + j][tx] = Wdec[(m0 + ty + j) * D_SAE + s0 + tx];
  __syncthreads();
#pragma unroll
  for (int j = 0; j < 32; j += 8)
    WdT[(s0 + ty + j) * D_MODEL + m0 + tx] = tile[tx][ty + j];
}

// ---------------------------------------------------------------------------
// Kernel 1 (primary): bf16 WMMA encoder, async-to-LDS double-buffered staging,
// manually unrolled x2 so buffer indices are compile-time and accumulators
// chain in place. pre_act stores are non-temporal (write-once stream; keep
// X16/W16 resident in L2 instead).
// ---------------------------------------------------------------------------
__global__ __launch_bounds__(256)
void sae_encoder_wmma_async(const unsigned short* __restrict__ X16,
                            const unsigned short* __restrict__ W16,
                            const float* __restrict__ benc,
                            float* __restrict__ pre) {
  __shared__ unsigned short As[2][128][40];   // padded: 80B row stride, conflict-free
  __shared__ unsigned short Bs[2][128][40];

  const int tid   = threadIdx.x;
  const int lane  = tid & 31;
  const int wid   = tid >> 5;
  const int waveM = wid >> 2;               // 0..1
  const int waveN = wid & 3;                // 0..3
  const int half  = lane >> 4;
  const int l16   = lane & 15;

  const int m0 = blockIdx.y * 128;
  const int n0 = blockIdx.x * 128;

  // staging map: 512 x 16B chunks per matrix per K-step; 2 chunks/thread/matrix
  const int rA = tid >> 2;                  // row 0..63 (and +64)
  const int c4 = tid & 3;                   // 16B chunk within 64B row segment
  const unsigned voff0 = (unsigned)(rA * (D_MODEL * 2) + c4 * 16);
  const unsigned voff1 = (unsigned)((rA + 64) * (D_MODEL * 2) + c4 * 16);
  unsigned ldsA0[2], ldsA1[2], ldsB0[2], ldsB1[2];
#pragma unroll
  for (int b = 0; b < 2; ++b) {
    ldsA0[b] = lds_off(&As[b][rA][c4 * 8]);
    ldsA1[b] = lds_off(&As[b][rA + 64][c4 * 8]);
    ldsB0[b] = lds_off(&Bs[b][rA][c4 * 8]);
    ldsB1[b] = lds_off(&Bs[b][rA + 64][c4 * 8]);
  }

  f32x8 acc[4][2];
  const f32x8 zero = {0.f, 0.f, 0.f, 0.f, 0.f, 0.f, 0.f, 0.f};
#pragma unroll
  for (int mt = 0; mt < 4; ++mt)
#pragma unroll
    for (int nt = 0; nt < 2; ++nt) acc[mt][nt] = zero;

#define STAGE(RR, BUF) do {                                                    \
    const unsigned short* sA_ = X16 + (size_t)m0 * D_MODEL + (RR) * 32;        \
    const unsigned short* sB_ = W16 + (size_t)n0 * D_MODEL + (RR) * 32;        \
    ASYNC_LOAD_B128(ldsA0[BUF], voff0, sA_);                                   \
    ASYNC_LOAD_B128(ldsA1[BUF], voff1, sA_);                                   \
    ASYNC_LOAD_B128(ldsB0[BUF], voff0, sB_);                                   \
    ASYNC_LOAD_B128(ldsB1[BUF], voff1, sB_);                                   \
  } while (0)

#define COMPUTE(BUF) do {                                                      \
    FragAB aF[4], bF[2];                                                       \
    _Pragma("unroll")                                                          \
    for (int mt = 0; mt < 4; ++mt) {                                           \
      const unsigned short* p = &As[BUF][waveM * 64 + mt * 16 + l16][0];       \
      aF[mt].u[0] = *(const uint4*)(p + half * 8);                             \
      aF[mt].u[1] = *(const uint4*)(p + 16 + half * 8);                        \
    }                                                                          \
    _Pragma("unroll")                                                          \
    for (int nt = 0; nt < 2; ++nt) {                                           \
      const unsigned short* p = &Bs[BUF][waveN * 32 + nt * 16 + l16][0];       \
      bF[nt].u[0] = *(const uint4*)(p + half * 8);                             \
      bF[nt].u[1] = *(const uint4*)(p + 16 + half * 8);                        \
    }                                                                          \
    _Pragma("unroll")                                                          \
    for (int mt = 0; mt < 4; ++mt)                                             \
      _Pragma("unroll")                                                        \
      for (int nt = 0; nt < 2; ++nt)                                           \
        acc[mt][nt] = __builtin_amdgcn_wmma_f32_16x16x32_bf16(                 \
            false, aF[mt].v, false, bF[nt].v, (short)0, acc[mt][nt],           \
            false, false);                                                     \
  } while (0)

  const int NSTEP = D_MODEL / 32;           // 24 (even)
  STAGE(0, 0);
  for (int r = 0; r < NSTEP; r += 2) {
    STAGE(r + 1, 1);                        // r+1 <= NSTEP-1 always
    WAIT_ASYNC(4);                          // in-order: stage r landed
    __syncthreads();
    COMPUTE(0);
    __syncthreads();                        // buf0 free for overwrite

    if (r + 2 < NSTEP) { STAGE(r + 2, 0); WAIT_ASYNC(4); }
    else               { WAIT_ASYNC(0); }
    __syncthreads();
    COMPUTE(1);
    __syncthreads();                        // buf1 free for overwrite
  }
#undef STAGE
#undef COMPUTE

#pragma unroll
  for (int nt = 0; nt < 2; ++nt) {
    const int col  = n0 + waveN * 32 + nt * 16 + l16;
    const float bias = benc[col];
#pragma unroll
    for (int mt = 0; mt < 4; ++mt) {
      const int rbase = m0 + waveM * 64 + mt * 16 + half * 8;
#pragma unroll
      for (int v = 0; v < 8; ++v) {
        const float o = acc[mt][nt][v] + bias;
        __builtin_nontemporal_store(o, &pre[(size_t)(rbase + v) * D_SAE + col]);
      }
    }
  }
}

// ---------------------------------------------------------------------------
// Kernel 1 (fallback, no workspace): f32 inputs, convert in-kernel.
// ---------------------------------------------------------------------------
__global__ __launch_bounds__(256)
void sae_encoder_wmma(const float* __restrict__ x,
                      const float* __restrict__ Wenc,
                      const float* __restrict__ benc,
                      float* __restrict__ pre) {
  __shared__ unsigned short As[128][40];
  __shared__ unsigned short Bs[128][40];

  const int tid   = threadIdx.x;
  const int lane  = tid & 31;
  const int wid   = tid >> 5;
  const int waveM = wid >> 2;
  const int waveN = wid & 3;
  const int half  = lane >> 4;
  const int l16   = lane & 15;

  const int m0 = blockIdx.y * 128;
  const int n0 = blockIdx.x * 128;

  f32x8 acc[4][2];
  const f32x8 zero = {0.f, 0.f, 0.f, 0.f, 0.f, 0.f, 0.f, 0.f};
#pragma unroll
  for (int mt = 0; mt < 4; ++mt)
#pragma unroll
    for (int nt = 0; nt < 2; ++nt) acc[mt][nt] = zero;

  const int ldRow = tid >> 3;
  const int ldC4  = tid & 7;

  for (int kk = 0; kk < D_MODEL; kk += 32) {
    __syncthreads();
#pragma unroll
    for (int i = 0; i < 4; ++i) {
      const int row = ldRow + i * 32;
      float4 va = *(const float4*)&x[(size_t)(m0 + row) * D_MODEL + kk + ldC4 * 4];
      ushort4 sa = { f2bf(va.x), f2bf(va.y), f2bf(va.z), f2bf(va.w) };
      *(ushort4*)&As[row][ldC4 * 4] = sa;
      float4 vb = *(const float4*)&Wenc[(size_t)(n0 + row) * D_MODEL + kk + ldC4 * 4];
      ushort4 sb = { f2bf(vb.x), f2bf(vb.y), f2bf(vb.z), f2bf(vb.w) };
      *(ushort4*)&Bs[row][ldC4 * 4] = sb;
    }
    __syncthreads();

    FragAB aF[4];
#pragma unroll
    for (int mt = 0; mt < 4; ++mt) {
      const unsigned short* p = &As[waveM * 64 + mt * 16 + l16][0];
      aF[mt].u[0] = *(const uint4*)(p + half * 8);
      aF[mt].u[1] = *(const uint4*)(p + 16 + half * 8);
    }
    FragAB bF[2];
#pragma unroll
    for (int nt = 0; nt < 2; ++nt) {
      const unsigned short* p = &Bs[waveN * 32 + nt * 16 + l16][0];
      bF[nt].u[0] = *(const uint4*)(p + half * 8);
      bF[nt].u[1] = *(const uint4*)(p + 16 + half * 8);
    }
#pragma unroll
    for (int mt = 0; mt < 4; ++mt)
#pragma unroll
      for (int nt = 0; nt < 2; ++nt)
        acc[mt][nt] = __builtin_amdgcn_wmma_f32_16x16x32_bf16(
            false, aF[mt].v, false, bF[nt].v, (short)0, acc[mt][nt], false, false);
  }

#pragma unroll
  for (int nt = 0; nt < 2; ++nt) {
    const int col  = n0 + waveN * 32 + nt * 16 + l16;
    const float bias = benc[col];
#pragma unroll
    for (int mt = 0; mt < 4; ++mt) {
      const int rbase = m0 + waveM * 64 + mt * 16 + half * 8;
#pragma unroll
      for (int v = 0; v < 8; ++v) {
        const float o = acc[mt][nt][v] + bias;
        __builtin_nontemporal_store(o, &pre[(size_t)(rbase + v) * D_SAE + col]);
      }
    }
  }
}

// ---------------------------------------------------------------------------
// Kernel 2: per-row exact top-32 (histogram radix-select on float keys),
//           in-place sparsification of latents, fused sparse decode.
// ---------------------------------------------------------------------------
__global__ __launch_bounds__(256)
void sae_topk_decode(const float* __restrict__ Wdec,
                     const float* __restrict__ WdT,
                     float* __restrict__ latents,
                     float* __restrict__ recon,
                     int useT) {
  __shared__ unsigned int hist[4096];
  __shared__ unsigned int part[256];
  __shared__ unsigned int candKey[CAND];
  __shared__ int   selIdx[TOPK];
  __shared__ float selVal[TOPK];
  __shared__ unsigned int s_bstar, s_above, s_ncand, s_nsel, s_thresh;

  const int tid = threadIdx.x;
  const int row = blockIdx.x;
  float* pre = latents + (size_t)row * D_SAE;

  for (int i = tid; i < 4096; i += 256) hist[i] = 0u;
  if (tid == 0) { s_ncand = 0u; s_nsel = 0u; }
  __syncthreads();

  for (int i = tid; i < D_SAE; i += 256)
    atomicAdd(&hist[fkey(pre[i]) >> 20], 1u);
  __syncthreads();

  unsigned int psum = 0;
  for (int j = 0; j < 16; ++j) psum += hist[tid * 16 + j];
  part[tid] = psum;
  __syncthreads();
  if (tid == 0) {
    unsigned int run = 0;
    for (int t = 255; t >= 0; --t) { unsigned int p = part[t]; part[t] = run; run += p; }
  }
  __syncthreads();
  {
    unsigned int above = part[tid];
    for (int j = 15; j >= 0; --j) {
      const unsigned int c = hist[tid * 16 + j];
      if (above < TOPK && above + c >= TOPK) { s_bstar = (unsigned)(tid * 16 + j); s_above = above; }
      above += c;
    }
  }
  __syncthreads();

  const unsigned int bstar = s_bstar;
  for (int i = tid; i < D_SAE; i += 256) {
    const unsigned int k = fkey(pre[i]);
    if ((k >> 20) == bstar) {
      const unsigned int p = atomicAdd(&s_ncand, 1u);
      if (p < CAND) candKey[p] = k;
    }
  }
  __syncthreads();
  if (tid == 0) {
    const int n = (int)(s_ncand < CAND ? s_ncand : CAND);
    const int r = TOPK - (int)s_above;
    unsigned int th = 0u;
    for (int it = 0; it < r; ++it) {
      unsigned int best = 0u; int bi = 0;
      for (int c = 0; c < n; ++c)
        if (candKey[c] > best) { best = candKey[c]; bi = c; }
      candKey[bi] = 0u;
      th = best;
    }
    s_thresh = th;
  }
  __syncthreads();

  const unsigned int th = s_thresh;
  for (int i = tid; i < D_SAE; i += 256) {
    const float v = pre[i];
    float o = 0.0f;
    if (fkey(v) >= th) {
      o = v;
      const unsigned int p = atomicAdd(&s_nsel, 1u);
      if (p < TOPK) { selIdx[p] = i; selVal[p] = v; }
    }
    __builtin_nontemporal_store(o, &pre[i]);   // latents never re-read: keep L2 for WdT
  }
  __syncthreads();

  const int nsel = (int)(s_nsel < TOPK ? s_nsel : TOPK);
  float a0 = 0.f, a1 = 0.f, a2 = 0.f;
  if (useT) {
    for (int k = 0; k < nsel; ++k) {
      const float* wp = WdT + (size_t)selIdx[k] * D_MODEL;
      const float v = selVal[k];
      a0 += v * wp[tid];
      a1 += v * wp[tid + 256];
      a2 += v * wp[tid + 512];
    }
  } else {
    for (int k = 0; k < nsel; ++k) {
      const int idx = selIdx[k];
      const float v = selVal[k];
      a0 += v * Wdec[(size_t)(tid      ) * D_SAE + idx];
      a1 += v * Wdec[(size_t)(tid + 256) * D_SAE + idx];
      a2 += v * Wdec[(size_t)(tid + 512) * D_SAE + idx];
    }
  }
  float* rr = recon + (size_t)row * D_MODEL;
  rr[tid]       = a0;
  rr[tid + 256] = a1;
  rr[tid + 512] = a2;
}

// ---------------------------------------------------------------------------
extern "C" void kernel_launch(void* const* d_in, const int* in_sizes, int n_in,
                              void* d_out, int out_size, void* d_ws, size_t ws_size,
                              hipStream_t stream) {
  const float* x    = (const float*)d_in[0];
  const float* Wenc = (const float*)d_in[1];
  const float* benc = (const float*)d_in[2];
  const float* Wdec = (const float*)d_in[3];

  float* recon   = (float*)d_out;                      // [8192, 768]
  float* latents = recon + (size_t)BATCH * D_MODEL;    // [8192, 24576]

  const size_t x16B = (size_t)BATCH * D_MODEL * sizeof(unsigned short);
  const size_t w16B = (size_t)D_SAE * D_MODEL * sizeof(unsigned short);
  const size_t wdtB = (size_t)D_SAE * D_MODEL * sizeof(float);

  char* ws = (char*)d_ws;
  unsigned short* X16 = nullptr;
  unsigned short* W16 = nullptr;
  float* WdT = nullptr;
  int haveAsync = 0, haveT = 0;

  if (ws && ws_size >= x16B + w16B) {
    haveAsync = 1;
    X16 = (unsigned short*)ws;
    W16 = (unsigned short*)(ws + x16B);
    if (ws_size >= x16B + w16B + wdtB) { haveT = 1; WdT = (float*)(ws + x16B + w16B); }
  } else if (ws && ws_size >= wdtB) {
    haveT = 1; WdT = (float*)ws;
  }

  if (haveT)
    wdec_transpose<<<dim3(D_SAE / 32, D_MODEL / 32), 256, 0, stream>>>(Wdec, WdT);

  if (haveAsync) {
    const int nx = BATCH * D_MODEL;
    const int nw = D_SAE * D_MODEL;
    cvt_bf16<<<dim3(nx / (256 * 8)), 256, 0, stream>>>(x, X16, nx);
    cvt_bf16<<<dim3(nw / (256 * 8)), 256, 0, stream>>>(Wenc, W16, nw);
    sae_encoder_wmma_async<<<dim3(D_SAE / 128, BATCH / 128), 256, 0, stream>>>(X16, W16, benc, latents);
  } else {
    sae_encoder_wmma<<<dim3(D_SAE / 128, BATCH / 128), 256, 0, stream>>>(x, Wenc, benc, latents);
  }

  sae_topk_decode<<<dim3(BATCH), 256, 0, stream>>>(Wdec, WdT, latents, recon, haveT);
}
